// TileInit_7550552506908
// MI455X (gfx1250) — matrix operand order repaired and checked by reading, not compiled, not run on hardware
//
#include <hip/hip_runtime.h>

// ---------------------------------------------------------------------------
// Types for CDNA5 WMMA bf16 path + TDM descriptors
// ---------------------------------------------------------------------------
typedef __bf16 bf16;
typedef __attribute__((ext_vector_type(16))) __bf16 v16bf;
typedef __attribute__((ext_vector_type(8)))  __bf16 v8bf;
typedef __attribute__((ext_vector_type(8)))  float  v8f;
typedef __attribute__((ext_vector_type(4)))  unsigned int u32x4;
typedef __attribute__((ext_vector_type(8)))  int i32x8;
typedef __attribute__((ext_vector_type(4)))  int i32x4;

// ---------------------------------------------------------------------------
// Problem constants
// ---------------------------------------------------------------------------
#define B_   8
#define C_   128
#define H_   64
#define W_   128
#define G_   8
#define CG_  16
#define D_   24
#define HW_  (H_ * W_)
#define DHW_ (D_ * H_ * W_)

// Zero-padded activation dims: index (z+1, y+1, x+1); z padded 3 past the end
// so the dead padded tap 27 (zero A-fragment) still reads in-bounds zeros.
#define DP_  27   // D_ + 3
#define HP_  66   // H_ + 2
#define WP_  130  // W_ + 2

// conv GEMM chunking (K ordered tap-major: k = tap*Cin + ci)
#define NCH1 7    // Cin=8 : 28 taps * 8   = 224 = 7*32
#define NCH2 14   // Cin=16: 28 taps * 16  = 448 = 14*32
#define NCHF 36   // Cin=128: 9 taps * 128 = 1152 = 36*32 (exact)

// Constant-folded tap -> padded-offset decode (t is a literal after unroll)
#define TKZ(t) ((t) / 9)
#define TKY(t) (((t) % 9) / 3)
#define TKX(t) ((t) % 3)
#define TOFF(t, Cc) ((((TKZ(t) * HP_) + TKY(t)) * WP_ + TKX(t)) * (Cc))
#define FOFF(t) (((((t) / 3) * WP_) + ((t) % 3)) * 128)

// ---------------------------------------------------------------------------
// TDM weight staging: one tensor_load_to_lds per block (issued by wave 0),
// flat nbytes copy expressed as a 1-row 2D tile of 8-byte elements.
// D# field packing per CDNA5 ISA ch.8 (group0 128b, group1 256b).
// This toolchain exposes the 6-arg builtin (extra descriptor group + cpol).
// ---------------------------------------------------------------------------
__device__ __forceinline__ void tdm_stage_weights(const bf16* __restrict__ gsrc,
                                                  bf16* lds_dst, int nbytes) {
    if (threadIdx.x < 32) {   // wave 0 only (wave-uniform branch)
        unsigned long long ga = (unsigned long long)gsrc;
        unsigned int la = (unsigned int)(unsigned long long)lds_dst; // LDS offset (addr[31:0])
        unsigned int n8 = (unsigned int)(nbytes >> 3);               // 8-byte elements
        u32x4 g0 = { 1u,                                  // count=1 (valid user D#)
                     la,                                  // lds_addr
                     (unsigned int)(ga & 0xFFFFFFFFu),    // global_addr[31:0]
                     (unsigned int)((ga >> 32) & 0x01FFFFFFu) | (2u << 30) }; // ga[56:32] | type=2
        i32x8 g1 = { (int)(3u << 16),                         // wg_mask=0, data_size=3 (8B)
                     (int)((n8 & 0xFFFFu) << 16),             // tensor_dim0[15:0] @ bits 63:48
                     (int)((n8 >> 16) | (1u << 16)),          // tensor_dim0[31:16], tensor_dim1=1
                     (int)((n8 & 0xFFFFu) << 16),             // tile_dim0 @ bits 127:112
                     1,                                       // tile_dim1=1, tile_dim2=0
                     (int)n8,                                 // tensor_dim0_stride[31:0]
                     0, 0 };
        i32x4 gz  = { 0, 0, 0, 0 };
        i32x8 gz8 = { 0, 0, 0, 0, 0, 0, 0, 0 };
        __builtin_amdgcn_tensor_load_to_lds(g0, g1, gz, gz, gz8, 0);
        __builtin_amdgcn_s_wait_tensorcnt((short)0);
    }
    __syncthreads();
}

// ---------------------------------------------------------------------------
// Zero-fill (16B per thread) for the padded halo buffers
// ---------------------------------------------------------------------------
__global__ void zero_v4(uint4* __restrict__ p, int n) {
    int i = blockIdx.x * blockDim.x + threadIdx.x;
    if (i < n) p[i] = make_uint4(0u, 0u, 0u, 0u);
}

// ---------------------------------------------------------------------------
// Pack conv weights straight into per-lane WMMA A-fragment order:
// wp[(ch*32 + lane)*16 + j], lane row m = lane&15, hi = lane>>4,
// element j <-> k = ch*32 + hi*16 + j, k = tap*Cin + ci.
// Source weights: w[(o*Cin + ci)*Ktaps + tap].
// ---------------------------------------------------------------------------
__global__ void pack_weights_frag(const float* __restrict__ w, bf16* __restrict__ wp,
                                  int lgC, int Ktaps, int nchunks) {
    int idx = blockIdx.x * blockDim.x + threadIdx.x;
    if (idx >= nchunks * 512) return;
    int j    = idx & 15;
    int lane = (idx >> 4) & 31;
    int ch   = idx >> 9;
    int m  = lane & 15, hi = lane >> 4;
    int kl = hi * 16 + j;
    int k  = ch * 32 + kl;
    int tap = k >> lgC;
    int ci  = k & ((1 << lgC) - 1);
    float v = (tap < Ktaps) ? w[((m << lgC) + ci) * Ktaps + tap] : 0.0f;
    wp[idx] = (bf16)v;
}

// fL (B,128,H,W) f32 -> padded channels-last (B,HP,WP,128) bf16
__global__ void transpose_fl_bf16(const float* __restrict__ x, bf16* __restrict__ y, int n) {
    int i = blockIdx.x * blockDim.x + threadIdx.x;
    if (i >= n) return;
    int c = i & 127;
    int w = (i >> 7) & 127;
    int h = (i >> 14) & 63;
    int b = i >> 20;
    y[(((size_t)b * HP_ + h + 1) * WP_ + w + 1) * 128 + c] =
        (bf16)x[((size_t)b * C_ + c) * HW_ + h * W_ + w];
}

// ---------------------------------------------------------------------------
// Group-correlation cost volume -> padded channels-last cv (B,DP,HP,WP,8) bf16
// ---------------------------------------------------------------------------
__global__ void cost_volume_cl(const float* __restrict__ fL, const float* __restrict__ fR,
                               bf16* __restrict__ cv) {
    int t = blockIdx.x * blockDim.x + threadIdx.x;
    if (t >= B_ * G_ * HW_) return;
    int w = t % W_;
    int h = (t / W_) % H_;
    int g = (t / HW_) % G_;
    int b = t / (G_ * HW_);

    const float* pL = fL + ((size_t)(b * C_ + g * CG_)) * HW_ + h * W_ + w;
    float l[CG_];
#pragma unroll
    for (int c = 0; c < CG_; ++c) l[c] = pL[c * HW_];

    const float* pRbase = fR + ((size_t)(b * C_ + g * CG_)) * HW_ + h * W_;
    __builtin_prefetch(pRbase + w, 0, 0);
    bf16* out = cv + (((size_t)b * DP_ + 1) * HP_ + h + 1) * WP_ * 8 + (w + 1) * 8 + g;
    for (int d = 0; d < D_; ++d) {
        float acc = 0.0f;
        if (w >= d) {
            const float* pR = pRbase + (w - d);
#pragma unroll
            for (int c = 0; c < CG_; ++c) acc += l[c] * pR[c * HW_];
        }
        out[(size_t)d * HP_ * WP_ * 8] = (bf16)(acc * (1.0f / CG_));
    }
}

// ---------------------------------------------------------------------------
// conv1: 3x3x3, Cin=8 -> 16. Unconditional 16B B-loads from padded input.
// Even/odd chunk accumulator chains avoid WMMA->WMMA RAW hazard NOPs.
// ---------------------------------------------------------------------------
__global__ __launch_bounds__(256) void conv3d_wmma_c8(
    const bf16* __restrict__ act, const bf16* __restrict__ wfrag,
    float* __restrict__ out) {
    __shared__ bf16 s_w[NCH1 * 512];
    tdm_stage_weights(wfrag, s_w, NCH1 * 1024);

    int lane = threadIdx.x & 31;
    int tile = blockIdx.x * 8 + (threadIdx.x >> 5);
    int wt = tile & 7;
    int h  = (tile >> 3) & 63;
    int d  = (tile >> 9) % D_;
    int b  = tile / (D_ << 9);
    int w0 = wt * 16;
    int n  = lane & 15, hi = lane >> 4;

    const bf16* p0 = act + (((size_t)(b * DP_ + d) * HP_ + h) * WP_ + w0 + n) * 8;

    v8f accE = {}, accO = {};
#pragma unroll
    for (int ch = 0; ch < NCH1; ++ch) {
        v16bf a = *(const v16bf*)&s_w[(ch * 32 + lane) * 16];
        const int olo0 = TOFF(4 * ch + 0, 8), ohi0 = TOFF(4 * ch + 1, 8);
        const int olo1 = TOFF(4 * ch + 2, 8), ohi1 = TOFF(4 * ch + 3, 8);
        int olo = hi ? olo1 : olo0;
        int ohi = hi ? ohi1 : ohi0;
        v8bf plo = *(const v8bf*)(p0 + olo);
        v8bf phi = *(const v8bf*)(p0 + ohi);
        v16bf bm = __builtin_shufflevector(plo, phi, 0, 1, 2, 3, 4, 5, 6, 7,
                                           8, 9, 10, 11, 12, 13, 14, 15);
        if (ch & 1)
            accO = __builtin_amdgcn_wmma_f32_16x16x32_bf16(
                false, a, false, bm, (short)0, accO, false, false);
        else
            accE = __builtin_amdgcn_wmma_f32_16x16x32_bf16(
                false, a, false, bm, (short)0, accE, false, false);
    }
    v8f acc = accE + accO;
    float* po = &out[(((size_t)(b * D_ + d) * H_ + h) * W_ + w0 + n) * 16 + hi * 8];
    *(v8f*)po = acc;
}

// ---------------------------------------------------------------------------
// conv2: 3x3x3, Cin=16 -> 16 (FLOP-dominant). Two N-tiles per wave: each A
// fragment feeds two WMMAs, and the alternating acc chains hide RAW hazards.
// ---------------------------------------------------------------------------
__global__ __launch_bounds__(256) void conv3d_wmma_c16(
    const bf16* __restrict__ act, const bf16* __restrict__ wfrag,
    float* __restrict__ out) {
    __shared__ bf16 s_w[NCH2 * 512];
    tdm_stage_weights(wfrag, s_w, NCH2 * 1024);

    int lane = threadIdx.x & 31;
    int tile = blockIdx.x * 8 + (threadIdx.x >> 5);   // B*D*H*(W/32) tiles
    int wt = tile & 3;
    int h  = (tile >> 2) & 63;
    int d  = (tile >> 8) % D_;
    int b  = tile / (D_ << 8);
    int w0 = wt * 32;
    int n  = lane & 15, hi = lane >> 4;

    const bf16* p0 = act + (((size_t)(b * DP_ + d) * HP_ + h) * WP_ + w0 + n) * 16;

    v8f acc0 = {}, acc1 = {};
#pragma unroll
    for (int ch = 0; ch < NCH2; ++ch) {
        v16bf a = *(const v16bf*)&s_w[(ch * 32 + lane) * 16];
        const int o0 = TOFF(2 * ch + 0, 16);
        const int o1 = TOFF(2 * ch + 1, 16);
        int off = hi ? o1 : o0;
        v16bf b0 = *(const v16bf*)(p0 + off);
        v16bf b1 = *(const v16bf*)(p0 + off + 16 * 16);
        acc0 = __builtin_amdgcn_wmma_f32_16x16x32_bf16(
            false, a, false, b0, (short)0, acc0, false, false);
        acc1 = __builtin_amdgcn_wmma_f32_16x16x32_bf16(
            false, a, false, b1, (short)0, acc1, false, false);
    }
    float* po = &out[(((size_t)(b * D_ + d) * H_ + h) * W_ + w0 + n) * 16 + hi * 8];
    *(v8f*)po = acc0;
    *(v8f*)(po + 16 * 16) = acc1;
}

// ---------------------------------------------------------------------------
// Feature head: 3x3 conv2d 128 -> 16. K = 9 taps * 128 (exact, all valid).
// tap = ch>>2 is hi-independent: pure constant offsets + hi*16.
// Even/odd chunk accumulator chains hide WMMA RAW hazards.
// ---------------------------------------------------------------------------
__global__ __launch_bounds__(256) void conv2d_fh_wmma(
    const bf16* __restrict__ act, const bf16* __restrict__ wfrag,
    float* __restrict__ out) {
    __shared__ bf16 s_w[NCHF * 512];   // 36 KB
    tdm_stage_weights(wfrag, s_w, NCHF * 1024);

    int lane = threadIdx.x & 31;
    int tile = blockIdx.x * 8 + (threadIdx.x >> 5);   // B*H*8 tiles
    int wt = tile & 7;
    int h  = (tile >> 3) & 63;
    int b  = tile >> 9;
    int w0 = wt * 16;
    int n  = lane & 15, hi = lane >> 4;

    const bf16* p0 = act + (((size_t)(b * HP_ + h) * WP_) + w0 + n) * 128 + hi * 16;

    v8f accE = {}, accO = {};
#pragma unroll
    for (int ch = 0; ch < NCHF; ++ch) {
        v16bf a = *(const v16bf*)&s_w[(ch * 32 + lane) * 16];
        const int off = FOFF(ch >> 2) + (ch & 3) * 32;
        v16bf bm = *(const v16bf*)(p0 + off);
        if (ch & 1)
            accO = __builtin_amdgcn_wmma_f32_16x16x32_bf16(
                false, a, false, bm, (short)0, accO, false, false);
        else
            accE = __builtin_amdgcn_wmma_f32_16x16x32_bf16(
                false, a, false, bm, (short)0, accE, false, false);
    }
    v8f acc = accE + accO;
    float* po = &out[((size_t)(b * H_ + h) * W_ + w0 + n) * 16 + hi * 8];
    *(v8f*)po = acc;
}

// ---------------------------------------------------------------------------
// GroupNorm on dense channels-last (..,16): C=16, 8 groups of 2 channels.
// ---------------------------------------------------------------------------
__global__ void gn_reduce_cl(const float* __restrict__ x, float* __restrict__ stats,
                             int spatial) {
    int bg = blockIdx.x;
    int b = bg >> 3, g = bg & 7;
    const float2* p = (const float2*)x;
    float s = 0.0f, sq = 0.0f;
    for (int i = threadIdx.x; i < spatial; i += blockDim.x) {
        float2 v = p[((size_t)b * spatial + i) * 8 + g];
        s  += v.x + v.y;
        sq += v.x * v.x + v.y * v.y;
    }
    __shared__ float ss[256], s2[256];
    int tid = threadIdx.x;
    ss[tid] = s; s2[tid] = sq;
    __syncthreads();
    for (int o = 128; o > 0; o >>= 1) {
        if (tid < o) { ss[tid] += ss[tid + o]; s2[tid] += s2[tid + o]; }
        __syncthreads();
    }
    if (tid == 0) {
        float inv = 1.0f / (2.0f * (float)spatial);
        float mu  = ss[0] * inv;
        float var = s2[0] * inv - mu * mu;
        stats[bg * 2]     = mu;
        stats[bg * 2 + 1] = rsqrtf(var + 1e-5f);
    }
}

// GN apply + SiLU: dense craw (B,D,H,W,16) f32 -> PADDED bf16 activations
__global__ void gn_apply_silu_pad(const float* __restrict__ x,
                                  const float* __restrict__ stats,
                                  const float* __restrict__ gw,
                                  const float* __restrict__ gb,
                                  bf16* __restrict__ y, int n) {
    int i = blockIdx.x * blockDim.x + threadIdx.x;
    if (i >= n) return;
    int c  = i & 15;
    int xw = (i >> 4) & 127;
    int yh = (i >> 11) & 63;
    int zd = (i >> 17) % D_;
    int b  = i / (D_ << 17);
    int bg = b * 8 + (c >> 1);
    float mu = stats[bg * 2], rs = stats[bg * 2 + 1];
    float v = (x[i] - mu) * rs * gw[c] + gb[c];
    y[(((size_t)(b * DP_ + zd + 1) * HP_ + yh + 1) * WP_ + xw + 1) * 16 + c] =
        (bf16)(v / (1.0f + __expf(-v)));
}

// Feature-head GN apply + SiLU: dense (B,H,W,16) in -> reference (B,16,H,W) out
__global__ void gn_apply_silu_fh(const float* __restrict__ x,
                                 const float* __restrict__ stats,
                                 const float* __restrict__ gw,
                                 const float* __restrict__ gb,
                                 float* __restrict__ y, int n) {
    int i = blockIdx.x * blockDim.x + threadIdx.x;
    if (i >= n) return;
    int p = i & (HW_ - 1);
    int c = (i >> 13) & 15;
    int b = i >> 17;
    int bg = b * 8 + (c >> 1);
    float mu = stats[bg * 2], rs = stats[bg * 2 + 1];
    float v = (x[((size_t)b * HW_ + p) * 16 + c] - mu) * rs * gw[c] + gb[c];
    y[i] = v / (1.0f + __expf(-v));
}

// ---------------------------------------------------------------------------
// Final 16->1 conv3d on VALU; reads PADDED x2 => zero bounds checks.
// ---------------------------------------------------------------------------
__global__ void conv3_logits(const bf16* __restrict__ x2, const float* __restrict__ w3,
                             const float* __restrict__ b3, float* __restrict__ logits) {
    __shared__ float sw[27 * 16];
    for (int i = threadIdx.x; i < 27 * 16; i += blockDim.x)
        sw[i] = w3[(i & 15) * 27 + (i >> 4)];
    __syncthreads();

    int t = blockIdx.x * blockDim.x + threadIdx.x;
    if (t >= B_ * D_ * HW_) return;
    int w = t % W_;
    int h = (t / W_) % H_;
    int d = (t / HW_) % D_;
    int b = t / (D_ * HW_);

    const bf16* p0 = &x2[(((size_t)(b * DP_ + d) * HP_ + h) * WP_ + w) * 16];
    float acc = b3[0];
#pragma unroll
    for (int tap = 0; tap < 27; ++tap) {
        const bf16* p = p0 + TOFF(tap, 16);
        const float* pw = &sw[tap * 16];
#pragma unroll
        for (int ci = 0; ci < 16; ++ci) acc += (float)p[ci] * pw[ci];
    }
    logits[t] = acc;
}

// Softmax over D per pixel: expected disparity, confidence, sx=sy=0.
__global__ void softmax_disp(const float* __restrict__ logits,
                             float* __restrict__ d_out, float* __restrict__ sx,
                             float* __restrict__ sy, float* __restrict__ conf) {
    int t = blockIdx.x * blockDim.x + threadIdx.x;
    if (t >= B_ * HW_) return;
    int p = t % HW_;
    int b = t / HW_;
    const float* L = logits + (size_t)b * D_ * HW_ + p;

    float mx = -3.0e38f;
#pragma unroll
    for (int d = 0; d < D_; ++d) mx = fmaxf(mx, L[(size_t)d * HW_]);
    float se = 0.0f, ed = 0.0f, em = 0.0f;
#pragma unroll
    for (int d = 0; d < D_; ++d) {
        float e = __expf(L[(size_t)d * HW_] - mx);
        se += e; ed += e * (float)d; em = fmaxf(em, e);
    }
    float inv = 1.0f / se;
    d_out[t] = ed * inv;
    conf[t]  = em * inv;
    sx[t] = 0.0f;
    sy[t] = 0.0f;
}

// ---------------------------------------------------------------------------
// Host-side orchestration
// ---------------------------------------------------------------------------
extern "C" void kernel_launch(void* const* d_in, const int* in_sizes, int n_in,
                              void* d_out, int out_size, void* d_ws, size_t ws_size,
                              hipStream_t stream) {
    const float* fL      = (const float*)d_in[0];
    const float* fR      = (const float*)d_in[1];
    const float* w1      = (const float*)d_in[2];
    const float* gn1_w   = (const float*)d_in[3];
    const float* gn1_b   = (const float*)d_in[4];
    const float* w2      = (const float*)d_in[5];
    const float* gn2_w   = (const float*)d_in[6];
    const float* gn2_b   = (const float*)d_in[7];
    const float* w3      = (const float*)d_in[8];
    const float* b3      = (const float*)d_in[9];
    const float* fh_w    = (const float*)d_in[10];
    const float* fh_gn_w = (const float*)d_in[11];
    const float* fh_gn_b = (const float*)d_in[12];

    float* out = (float*)d_out;
    float* o_d    = out;                 // (B,1,H,W)   65536
    float* o_sx   = out + 65536;
    float* o_sy   = out + 131072;
    float* o_feat = out + 196608;        // (B,16,H,W)  1048576
    float* o_conf = out + 1245184;

    const int nX   = B_ * 16 * D_ * HW_;                // 25,165,824 (dense)
    const int nL   = B_ * D_ * HW_;                     // 1,572,864
    const int nFL  = B_ * C_ * HW_;                     // 8,388,608
    const int nF   = B_ * 16 * HW_;                     // 1,048,576
    const size_t nCVP = (size_t)B_ * DP_ * HP_ * WP_ * 8;    // padded cv
    const size_t nXP  = (size_t)B_ * DP_ * HP_ * WP_ * 16;   // padded x1/x2
    const size_t nFLP = (size_t)B_ * HP_ * WP_ * 128;        // padded fLb

    // Workspace carve
    char* ws = (char*)d_ws;
    size_t off = 0;
    auto take = [&](size_t bytes) { char* p = ws + off; off += (bytes + 255) & ~size_t(255); return p; };
    bf16*  cvp    = (bf16*) take(nCVP * 2);          // (B,DP,HP,WP,8)
    float* craw   = (float*)take((size_t)nX * 4);    // dense (B,D,H,W,16), reused
    bf16*  x1p    = (bf16*) take(nXP * 2);           // padded (B,DP,HP,WP,16)
    bf16*  x2p    = (bf16*) take(nXP * 2);
    float* logits = (float*)take((size_t)nL * 4);
    bf16*  fLbp   = (bf16*) take(nFLP * 2);          // padded (B,HP,WP,128)
    float* fraw   = (float*)take((size_t)nF * 4);    // dense (B,H,W,16)
    bf16*  wp1    = (bf16*) take((size_t)NCH1 * 512 * 2);
    bf16*  wp2    = (bf16*) take((size_t)NCH2 * 512 * 2);
    bf16*  wpf    = (bf16*) take((size_t)NCHF * 512 * 2);
    float* stats  = (float*)take(64 * 2 * 4);
    (void)ws_size; (void)in_sizes; (void)n_in; (void)out_size;

    const int TB = 256;

    // Zero the padded halo buffers (deterministic every call; ~7 us at 23 TB/s)
    {
        int n16;
        n16 = (int)(nCVP * 2 / 16);
        zero_v4<<<(n16 + TB - 1) / TB, TB, 0, stream>>>((uint4*)cvp, n16);
        n16 = (int)(nXP * 2 / 16);
        zero_v4<<<(n16 + TB - 1) / TB, TB, 0, stream>>>((uint4*)x1p, n16);
        zero_v4<<<(n16 + TB - 1) / TB, TB, 0, stream>>>((uint4*)x2p, n16);
        n16 = (int)(nFLP * 2 / 16);
        zero_v4<<<(n16 + TB - 1) / TB, TB, 0, stream>>>((uint4*)fLbp, n16);
    }

    // Weight packing (A-fragment layout) + fL padded channels-last downconvert
    pack_weights_frag<<<(NCH1 * 512 + TB - 1) / TB, TB, 0, stream>>>(w1, wp1, 3, 27, NCH1);
    pack_weights_frag<<<(NCH2 * 512 + TB - 1) / TB, TB, 0, stream>>>(w2, wp2, 4, 27, NCH2);
    pack_weights_frag<<<(NCHF * 512 + TB - 1) / TB, TB, 0, stream>>>(fh_w, wpf, 7, 9, NCHF);
    transpose_fl_bf16<<<(nFL + TB - 1) / TB, TB, 0, stream>>>(fL, fLbp, nFL);

    // Cost volume (padded channels-last)
    cost_volume_cl<<<(B_ * G_ * HW_) / TB, TB, 0, stream>>>(fL, fR, cvp);

    // conv1 (8->16) + GN + SiLU
    conv3d_wmma_c8<<<(B_ * D_ * H_ * (W_ / 16)) / 8, TB, 0, stream>>>(cvp, wp1, craw);
    gn_reduce_cl<<<64, TB, 0, stream>>>(craw, stats, DHW_);
    gn_apply_silu_pad<<<(nX + TB - 1) / TB, TB, 0, stream>>>(craw, stats, gn1_w, gn1_b, x1p, nX);

    // conv2 (16->16), 2 N-tiles per wave, + GN + SiLU
    conv3d_wmma_c16<<<(B_ * D_ * H_ * (W_ / 32)) / 8, TB, 0, stream>>>(x1p, wp2, craw);
    gn_reduce_cl<<<64, TB, 0, stream>>>(craw, stats, DHW_);
    gn_apply_silu_pad<<<(nX + TB - 1) / TB, TB, 0, stream>>>(craw, stats, gn2_w, gn2_b, x2p, nX);

    // logits + softmax head
    conv3_logits<<<(nL + TB - 1) / TB, TB, 0, stream>>>(x2p, w3, b3, logits);
    softmax_disp<<<(B_ * HW_) / TB, TB, 0, stream>>>(logits, o_d, o_sx, o_sy, o_conf);

    // feature head: conv2d 128->16 + GN + SiLU (-> (B,16,H,W) in d_out)
    conv2d_fh_wmma<<<(B_ * H_ * (W_ / 16)) / 8, TB, 0, stream>>>(fLbp, wpf, fraw);
    gn_reduce_cl<<<64, TB, 0, stream>>>(fraw, stats, HW_);
    gn_apply_silu_fh<<<(nF + TB - 1) / TB, TB, 0, stream>>>(fraw, stats, fh_gn_w, fh_gn_b, o_feat, nF);
}
